// _RPN_86878598463652
// MI455X (gfx1250) — matrix operand order repaired and checked by reading, not compiled, not run on hardware
//
#include <hip/hip_runtime.h>

typedef __attribute__((ext_vector_type(16))) __bf16 v16bf;
typedef __attribute__((ext_vector_type(8)))  float  v8f;

union Frag { v16bf v; uint4 q[2]; };

#define BATCH  8
#define CIN    512
#define COUT   512
#define Hh     50
#define Ww     76
#define HW     3800          // 50*76
#define MT     128           // Cout tile
#define NT     64            // pixel tile
#define KC     32            // K per WMMA chunk
#define NCHUNK (CIN / KC)    // 16
#define PAD    77            // max |flat shift| for a 3x3 tap = 76+1
#define XWIN   224           // NT + 2*PAD = 218, padded
#define XR     40            // LDS row stride (bf16) for X, 80B -> 16B aligned rows
#define WR     40            // LDS row stride (bf16) for W

// ---------------------------------------------------------------------------
// Kernel 0: split conv weights f32 -> bf16 hi/lo, layout [tap][ci][co]
// ---------------------------------------------------------------------------
__global__ __launch_bounds__(256)
void rpn_prep_weights(const float* __restrict__ w,
                      __bf16* __restrict__ whi, __bf16* __restrict__ wlo) {
    int e = blockIdx.x * 256 + threadIdx.x;          // e = (tap*512 + ci)*512 + co
    if (e >= 9 * CIN * COUT) return;
    int co  = e & 511;
    int ci  = (e >> 9) & 511;
    int tap = e >> 18;
    float v = w[((size_t)co * CIN + ci) * 9 + tap];
    __bf16 h = (__bf16)v;
    whi[e] = h;
    wlo[e] = (__bf16)(v - (float)h);
}

// ---------------------------------------------------------------------------
// Kernel 1: 3x3 conv + bias + ReLU as implicit GEMM with bf16x3 WMMA
// grid = (60 pixel tiles, 4 Cout tiles, 8 batches), block = 256 (8 waves)
// ---------------------------------------------------------------------------
__global__ __launch_bounds__(256)
void rpn_conv3x3_wmma(const float* __restrict__ x,
                      const __bf16* __restrict__ whi_t,
                      const __bf16* __restrict__ wlo_t,
                      const float* __restrict__ bias,
                      float* __restrict__ y) {
    __shared__ __align__(16) __bf16 sXhi[XWIN * XR];
    __shared__ __align__(16) __bf16 sXlo[XWIN * XR];
    __shared__ __align__(16) __bf16 sWhi[MT * WR];
    __shared__ __align__(16) __bf16 sWlo[MT * WR];

    const int tid  = threadIdx.x;
    const int lane = tid & 31;
    const int wv   = tid >> 5;
    const int half = lane >> 4;     // 0/1: lane group
    const int lm   = lane & 15;
    const int wm   = wv & 3;        // wave's 32-row Cout block within 128
    const int wn   = wv >> 2;       // wave's 32-col pixel block within 64

    const int n0      = blockIdx.x * NT;      // flat pixel base within batch
    const int co_base = blockIdx.y * MT;
    const int b       = blockIdx.z;

    int  nn[2], yy0[2], xx0[2], nl[2];
    bool inb[2];
#pragma unroll
    for (int ni = 0; ni < 2; ++ni) {
        nl[ni]  = wn * 32 + ni * 16 + lm;     // pixel within tile (B-frag column)
        nn[ni]  = n0 + nl[ni];
        yy0[ni] = nn[ni] / Ww;
        xx0[ni] = nn[ni] % Ww;
        inb[ni] = nn[ni] < HW;
    }

    v8f zero = {};
    v8f acc[2][2];
#pragma unroll
    for (int mi = 0; mi < 2; ++mi)
#pragma unroll
        for (int ni = 0; ni < 2; ++ni) acc[mi][ni] = zero;

#pragma unroll 1
    for (int ch = 0; ch < NCHUNK; ++ch) {
        const int ci0 = ch * KC;
#pragma unroll 1
        for (int tap = 0; tap < 9; ++tap) {
            if (ch | tap) __syncthreads();    // all LDS readers of prev stage done

            if (tap == 0) {
                // stage input window [n0-77, n0+147) for 32 ci, as bf16 hi/lo,
                // LDS layout X[pixel][ci] so B-frags read contiguous ci
                const float* xb = x + ((size_t)b * CIN + ci0) * HW;
#pragma unroll 1
                for (int i = 0; i < (XWIN * KC) / 256; ++i) {   // 28 iters
                    int e  = tid + 256 * i;
                    int ci = e / XWIN;
                    int pl = e - ci * XWIN;
                    int pw = n0 - PAD + pl;
                    float v = 0.0f;
                    if (pw >= 0 && pw < HW) v = xb[(size_t)ci * HW + pw];
                    __bf16 h = (__bf16)v;
                    sXhi[pl * XR + ci] = h;
                    sXlo[pl * XR + ci] = (__bf16)(v - (float)h);
                }
            }
            {   // stage this tap's weights: W[co][ci], coalesced over co
                const __bf16* wh = whi_t + ((size_t)tap * CIN + ci0) * COUT + co_base;
                const __bf16* wl = wlo_t + ((size_t)tap * CIN + ci0) * COUT + co_base;
#pragma unroll 1
                for (int i = 0; i < (MT * KC) / 256; ++i) {     // 16 iters
                    int e  = tid + 256 * i;
                    int co = e & (MT - 1);
                    int ci = e >> 7;
                    sWhi[co * WR + ci] = wh[(size_t)ci * COUT + co];
                    sWlo[co * WR + ci] = wl[(size_t)ci * COUT + co];
                }
            }
            __syncthreads();

            // A fragments: lane row M=lm; K runs {half*8..+7} and {16+half*8..+7}
            Frag ahi[2], alo[2];
#pragma unroll
            for (int mi = 0; mi < 2; ++mi) {
                int ml = wm * 32 + mi * 16 + lm;
                int o1 = ml * WR + half * 8;
                ahi[mi].q[0] = *(const uint4*)&sWhi[o1];
                ahi[mi].q[1] = *(const uint4*)&sWhi[o1 + 16];
                alo[mi].q[0] = *(const uint4*)&sWlo[o1];
                alo[mi].q[1] = *(const uint4*)&sWlo[o1 + 16];
            }

            // B fragments: lane col N=lm, K runs {half*8..+7} and {16+half*8..+7}; zero if the
            // shifted tap pixel is outside the image (per-lane, uniform over K)
            const int dy = tap / 3, dx = tap - dy * 3;
            const int shift = (dy - 1) * Ww + (dx - 1);
            Frag bhi[2], blo[2];
#pragma unroll
            for (int ni = 0; ni < 2; ++ni) {
                int xx = xx0[ni] + dx - 1;
                int yy = yy0[ni] + dy - 1;
                bool valid = inb[ni] && (unsigned)xx < (unsigned)Ww
                                     && (unsigned)yy < (unsigned)Hh;
                if (valid) {
                    int o = (nl[ni] + PAD + shift) * XR + half * 8;
                    bhi[ni].q[0] = *(const uint4*)&sXhi[o];
                    bhi[ni].q[1] = *(const uint4*)&sXhi[o + 16];
                    blo[ni].q[0] = *(const uint4*)&sXlo[o];
                    blo[ni].q[1] = *(const uint4*)&sXlo[o + 16];
                } else {
                    uint4 z = make_uint4(0u, 0u, 0u, 0u);
                    bhi[ni].q[0] = z; bhi[ni].q[1] = z;
                    blo[ni].q[0] = z; blo[ni].q[1] = z;
                }
            }

            // bf16x3: hi*hi + hi*lo + lo*hi, f32 accumulate
#pragma unroll
            for (int mi = 0; mi < 2; ++mi)
#pragma unroll
                for (int ni = 0; ni < 2; ++ni) {
                    acc[mi][ni] = __builtin_amdgcn_wmma_f32_16x16x32_bf16(
                        false, ahi[mi].v, false, bhi[ni].v, (short)0, acc[mi][ni], false, false);
                    acc[mi][ni] = __builtin_amdgcn_wmma_f32_16x16x32_bf16(
                        false, ahi[mi].v, false, blo[ni].v, (short)0, acc[mi][ni], false, false);
                    acc[mi][ni] = __builtin_amdgcn_wmma_f32_16x16x32_bf16(
                        false, alo[mi].v, false, bhi[ni].v, (short)0, acc[mi][ni], false, false);
                }
        }
    }

    // C/D layout: lane holds N=lm, rows M = half*8 + i in VGPR i
#pragma unroll
    for (int mi = 0; mi < 2; ++mi)
#pragma unroll
        for (int ni = 0; ni < 2; ++ni) {
            if (nn[ni] < HW) {
#pragma unroll
                for (int i = 0; i < 8; ++i) {
                    int co  = co_base + wm * 32 + mi * 16 + half * 8 + i;
                    float v = acc[mi][ni][i] + bias[co];
                    y[((size_t)b * COUT + co) * HW + nn[ni]] = fmaxf(v, 0.0f);
                }
            }
        }
}

// ---------------------------------------------------------------------------
// Kernel 2: 1x1 cls/bbox heads + paired softmax, one thread per pixel
// ---------------------------------------------------------------------------
__global__ __launch_bounds__(256)
void rpn_heads(const float* __restrict__ feat,
               const float* __restrict__ cls_w,  const float* __restrict__ cls_b,
               const float* __restrict__ bbox_w, const float* __restrict__ bbox_b,
               float* __restrict__ out) {
    __shared__ float sw[54 * 64];
    const int tid = threadIdx.x;
    const int g   = blockIdx.x * 256 + tid;
    const bool valid = g < BATCH * HW;
    const int b = g / HW;
    const int p = g - b * HW;

    float acc[54];
#pragma unroll
    for (int c = 0; c < 54; ++c) acc[c] = 0.0f;

#pragma unroll 1
    for (int kc = 0; kc < 8; ++kc) {
        const int k0 = kc * 64;
        if (kc) __syncthreads();
#pragma unroll 1
        for (int i = 0; i < 14; ++i) {           // 54*64 = 3456 weights
            int e = tid + 256 * i;
            if (e < 3456) {
                int co = e >> 6, k = e & 63;
                sw[e] = (co < 18) ? cls_w[(size_t)co * CIN + k0 + k]
                                  : bbox_w[(size_t)(co - 18) * CIN + k0 + k];
            }
        }
        __syncthreads();
        if (valid) {
            const float* fp = feat + ((size_t)b * CIN + k0) * HW + p;
#pragma unroll 1
            for (int k = 0; k < 64; ++k) {
                float f = fp[(size_t)k * HW];
#pragma unroll
                for (int c = 0; c < 54; ++c) acc[c] += f * sw[c * 64 + k];
            }
        }
    }

    if (valid) {
        // paired 2-way softmax over channels (a, a+9)
#pragma unroll
        for (int a = 0; a < 9; ++a) {
            float s0 = acc[a] + cls_b[a];
            float s1 = acc[a + 9] + cls_b[a + 9];
            float m  = fmaxf(s0, s1);
            float e0 = __expf(s0 - m), e1 = __expf(s1 - m);
            float inv = 1.0f / (e0 + e1);
            out[((size_t)b * 18 + a)     * HW + p] = e0 * inv;
            out[((size_t)b * 18 + a + 9) * HW + p] = e1 * inv;
        }
        float* ob = out + (size_t)BATCH * 18 * HW;
#pragma unroll
        for (int c = 0; c < 36; ++c)
            ob[((size_t)b * 36 + c) * HW + p] = acc[18 + c] + bbox_b[c];
    }
}

// ---------------------------------------------------------------------------
extern "C" void kernel_launch(void* const* d_in, const int* in_sizes, int n_in,
                              void* d_out, int out_size, void* d_ws, size_t ws_size,
                              hipStream_t stream) {
    const float* base_feat = (const float*)d_in[0];
    // d_in[1..3] = im_info, gt_boxes, num_boxes (unused by the reference outputs)
    const float* conv_w = (const float*)d_in[4];
    const float* conv_b = (const float*)d_in[5];
    const float* cls_w  = (const float*)d_in[6];
    const float* cls_b  = (const float*)d_in[7];
    const float* bbox_w = (const float*)d_in[8];
    const float* bbox_b = (const float*)d_in[9];
    float* out = (float*)d_out;

    char* ws = (char*)d_ws;
    float* conv_out = (float*)ws;                              // 8*512*3800 f32 = 62.26 MB
    const size_t CONV_BYTES = (size_t)BATCH * COUT * HW * sizeof(float);
    __bf16* whi = (__bf16*)(ws + CONV_BYTES);                  // 9*512*512 bf16
    __bf16* wlo = whi + (size_t)9 * CIN * COUT;

    rpn_prep_weights<<<(9 * CIN * COUT) / 256, 256, 0, stream>>>(conv_w, whi, wlo);

    dim3 grid((HW + NT - 1) / NT, COUT / MT, BATCH);           // (60, 4, 8)
    rpn_conv3x3_wmma<<<grid, 256, 0, stream>>>(base_feat, whi, wlo, conv_b, conv_out);

    rpn_heads<<<(BATCH * HW + 255) / 256, 256, 0, stream>>>(
        conv_out, cls_w, cls_b, bbox_w, bbox_b, out);
}
